// MultiHeadAttention_50062138802212
// MI455X (gfx1250) — matrix operand order repaired and checked by reading, not compiled, not run on hardware
//
#include <hip/hip_runtime.h>
#include <hip/hip_bf16.h>

// ---------------- problem constants ----------------
constexpr int Bc = 128;   // batch
constexpr int Tc = 256;   // sequence length
constexpr int Ec = 384;   // embed
constexpr int Hc = 6;     // heads
constexpr int Dc = 64;    // head dim
constexpr int TB = 64;    // t-rows per block (4 waves x 16)

typedef unsigned int  u32;
typedef unsigned short u16;
typedef __attribute__((ext_vector_type(16))) __bf16 v16bf;
typedef __attribute__((ext_vector_type(8)))  float  v8f;
typedef __attribute__((ext_vector_type(4)))  u32    u32x4;

// ---------------- helpers ----------------
__device__ __forceinline__ u16 f2bf(float x) {
    u32 u = __builtin_bit_cast(u32, x);
    u32 r = (u + 0x7FFFu + ((u >> 16) & 1u)) >> 16;   // round-to-nearest-even
    return (u16)r;
}
__device__ __forceinline__ u32 pack2bf(float a, float b) {
    return (u32)f2bf(a) | ((u32)f2bf(b) << 16);
}
__device__ __forceinline__ u32x4 ld16(const u16* p) {
    return *(const u32x4*)p;                          // 16B chunk (global or LDS b128)
}
__device__ __forceinline__ v16bf make_frag(u32x4 lo, u32x4 hi) {
    union { u32x4 q[2]; v16bf f; } u;
    u.q[0] = lo; u.q[1] = hi;
    return u.f;
}
__device__ __forceinline__ v8f zero8() {
    v8f z = {0.f,0.f,0.f,0.f,0.f,0.f,0.f,0.f};
    return z;
}
__device__ __forceinline__ v8f wmma_bf16(v16bf a, v16bf b, v8f c) {
    return __builtin_amdgcn_wmma_f32_16x16x32_bf16(false, a, false, b, (short)0, c, false, false);
}

// ---------------- kernel 0: weight transpose/convert ----------------
// Wq/Wk/Wv [H,E,D] f32 -> wqt/wkt/wvt [H,D,E] bf16 ; Wp [H*D,E] f32 -> wpt [E,H*D] bf16
__global__ __launch_bounds__(256) void prep_weights(
    const float* __restrict__ Wq, const float* __restrict__ Wk,
    const float* __restrict__ Wv, const float* __restrict__ Wp,
    u16* __restrict__ wqt, u16* __restrict__ wkt,
    u16* __restrict__ wvt, u16* __restrict__ wpt)
{
    int i = blockIdx.x * blockDim.x + threadIdx.x;   // [0, H*E*D) == [0, 384*384)
    if (i >= Hc * Ec * Dc) return;
    int h = i / (Ec * Dc);
    int r = i % (Ec * Dc);
    int e = r / Dc, d = r % Dc;
    int dst = (h * Dc + d) * Ec + e;
    wqt[dst] = f2bf(Wq[i]);
    wkt[dst] = f2bf(Wk[i]);
    wvt[dst] = f2bf(Wv[i]);
    int hd = i / Ec, e2 = i % Ec;                    // Wp[hd][e2] -> wpt[e2][hd]
    wpt[e2 * (Hc * Dc) + hd] = f2bf(Wp[i]);
}

// ---------------- kernel 1: QKV projection ----------------
// q,k: [B,H,T,D] bf16 ; vt: [B,H,D,T] bf16 (transposed for P@V)
__global__ __launch_bounds__(128) void qkv_kernel(
    const float* __restrict__ x,
    const float* __restrict__ bq, const float* __restrict__ bk, const float* __restrict__ bv,
    const u16* __restrict__ wqt, const u16* __restrict__ wkt, const u16* __restrict__ wvt,
    u16* __restrict__ qo, u16* __restrict__ ko, u16* __restrict__ vto)
{
    __shared__ u32 xs[TB * Ec / 2];                  // 64x384 bf16 = 48KB
    int b  = blockIdx.x / (Tc / TB);
    int tb = blockIdx.x % (Tc / TB);

    const float* xb = x + ((size_t)b * Tc + tb * TB) * Ec;
    for (int i = threadIdx.x; i < TB * Ec / 2; i += 128) {
        float2 v = ((const float2*)xb)[i];
        xs[i] = pack2bf(v.x, v.y);
    }
    __syncthreads();

    int lane = threadIdx.x & 31, wave = threadIdx.x >> 5;
    int m = lane & 15, hi = lane >> 4;
    int r0 = wave * 16;

    // preload 12 A-fragments (x rows, K = E in chunks of 32)
    v16bf a[12];
    const u16* xrow = (const u16*)xs + (r0 + m) * Ec;
    #pragma unroll
    for (int kb = 0; kb < 12; ++kb)
        a[kb] = make_frag(ld16(xrow + kb * 32 + hi * 8),
                          ld16(xrow + kb * 32 + 16 + hi * 8));

    for (int h = 0; h < Hc; ++h) {
        #pragma unroll
        for (int mat = 0; mat < 3; ++mat) {
            const u16*  wt   = (mat == 0 ? wqt : mat == 1 ? wkt : wvt) + (size_t)h * Dc * Ec;
            const float* bias = (mat == 0 ? bq : mat == 1 ? bk : bv) + h * Dc;
            #pragma unroll
            for (int db = 0; db < 4; ++db) {
                v8f acc = zero8();
                const u16* wrow = wt + (size_t)(db * 16 + m) * Ec;   // B lane: row N
                #pragma unroll
                for (int kb = 0; kb < 12; ++kb) {
                    v16bf bfrag = make_frag(ld16(wrow + kb * 32 + hi * 8),
                                            ld16(wrow + kb * 32 + 16 + hi * 8));
                    acc = wmma_bf16(a[kb], bfrag, acc);
                }
                float bb = bias[db * 16 + m];
                if (mat < 2) {
                    u16* out = (mat == 0 ? qo : ko)
                             + (((size_t)b * Hc + h) * Tc + tb * TB + r0) * Dc + db * 16 + m;
                    #pragma unroll
                    for (int r = 0; r < 8; ++r)
                        out[(size_t)(r + 8 * hi) * Dc] = f2bf(acc[r] + bb);
                } else {
                    u16* out = vto + (((size_t)b * Hc + h) * Dc + db * 16 + m) * Tc
                             + tb * TB + r0;
                    #pragma unroll
                    for (int r = 0; r < 8; ++r)
                        out[r + 8 * hi] = f2bf(acc[r] + bb);
                }
            }
        }
    }
}

// ---------------- kernel 2: causal attention per (b,h,64-query block) ----------------
__global__ __launch_bounds__(128) void attn_kernel(
    const u16* __restrict__ q, const u16* __restrict__ k, const u16* __restrict__ vt,
    u16* __restrict__ o)   // o: [B,T, H*D] bf16 (head-concatenated)
{
    __shared__ u16 pl[4 * 16 * Tc];                  // per-wave P stripe, 32KB
    int qb = blockIdx.x & 3;
    int h  = (blockIdx.x >> 2) % Hc;
    int b  = blockIdx.x / (4 * Hc);
    int lane = threadIdx.x & 31, wave = threadIdx.x >> 5;
    int m = lane & 15, hi = lane >> 4;
    int r0 = qb * 64 + wave * 16;                    // global query-row base of this wave

    const u16* qbase = q  + (((size_t)b * Hc + h) * Tc) * Dc;
    const u16* kbase = k  + (((size_t)b * Hc + h) * Tc) * Dc;
    const u16* vbase = vt + (((size_t)b * Hc + h) * Dc) * Tc;

    // Q A-fragments (row r0+m, d 0-31 and 32-63)
    const u16* qrow = qbase + (size_t)(r0 + m) * Dc;
    v16bf qa0 = make_frag(ld16(qrow + hi * 8),      ld16(qrow + 16 + hi * 8));
    v16bf qa1 = make_frag(ld16(qrow + 32 + hi * 8), ld16(qrow + 48 + hi * 8));

    int nbmax = qb * 4 + wave;                       // last live 16-key tile (causal)
    v8f s[16];

    // S = Q @ K^T, tile by tile, with causal mask
    #pragma unroll
    for (int nb = 0; nb < 16; ++nb) {
        if (nb <= nbmax) {
            const u16* krow = kbase + (size_t)(nb * 16 + m) * Dc;   // B lane: key row N
            v16bf kb0 = make_frag(ld16(krow + hi * 8),      ld16(krow + 16 + hi * 8));
            v16bf kb1 = make_frag(ld16(krow + 32 + hi * 8), ld16(krow + 48 + hi * 8));
            v8f c = zero8();
            c = wmma_bf16(qa0, kb0, c);
            c = wmma_bf16(qa1, kb1, c);
            int col = nb * 16 + m;
            #pragma unroll
            for (int r = 0; r < 8; ++r) {
                int row = r0 + r + 8 * hi;
                c[r] = (col > row) ? -3.0e38f : c[r];
            }
            s[nb] = c;
        }
    }

    // rowwise softmax (rows live across 16-lane halves; reduce with xor shuffles)
    float mx[8], sum[8];
    #pragma unroll
    for (int r = 0; r < 8; ++r) mx[r] = -3.0e38f;
    #pragma unroll
    for (int nb = 0; nb < 16; ++nb)
        if (nb <= nbmax)
            #pragma unroll
            for (int r = 0; r < 8; ++r) mx[r] = fmaxf(mx[r], s[nb][r]);
    #pragma unroll
    for (int r = 0; r < 8; ++r) {
        mx[r] = fmaxf(mx[r], __shfl_xor(mx[r], 1, 32));
        mx[r] = fmaxf(mx[r], __shfl_xor(mx[r], 2, 32));
        mx[r] = fmaxf(mx[r], __shfl_xor(mx[r], 4, 32));
        mx[r] = fmaxf(mx[r], __shfl_xor(mx[r], 8, 32));
    }
    #pragma unroll
    for (int r = 0; r < 8; ++r) sum[r] = 0.f;
    #pragma unroll
    for (int nb = 0; nb < 16; ++nb)
        if (nb <= nbmax)
            #pragma unroll
            for (int r = 0; r < 8; ++r) {
                float e = __expf(s[nb][r] - mx[r]);
                s[nb][r] = e;
                sum[r] += e;
            }
    #pragma unroll
    for (int r = 0; r < 8; ++r) {
        sum[r] += __shfl_xor(sum[r], 1, 32);
        sum[r] += __shfl_xor(sum[r], 2, 32);
        sum[r] += __shfl_xor(sum[r], 4, 32);
        sum[r] += __shfl_xor(sum[r], 8, 32);
        sum[r] = 1.0f / sum[r];
    }

    // write normalized P (bf16) into this wave's LDS stripe [16 x 256] row-major
    u16* pw = pl + wave * 16 * Tc;
    #pragma unroll
    for (int nb = 0; nb < 16; ++nb)
        if (nb <= nbmax)
            #pragma unroll
            for (int r = 0; r < 8; ++r)
                pw[(r + 8 * hi) * Tc + nb * 16 + m] = f2bf(s[nb][r] * sum[r]);
    if ((nbmax & 1) == 0) {                          // zero the dead half of the last 32-key block
        int nz = nbmax + 1;
        #pragma unroll
        for (int r = 0; r < 8; ++r)
            pw[(r + 8 * hi) * Tc + nz * 16 + m] = 0;
    }

    // O = P @ V^T  (K = keys, in 32-chunks)
    int kbmax = nbmax >> 1;
    const u16* prow = pw + m * Tc;                   // A lane: row m of P
    v16bf pa[8];
    #pragma unroll
    for (int kb = 0; kb < 8; ++kb)
        if (kb <= kbmax)
            pa[kb] = make_frag(ld16(prow + kb * 32 + hi * 8),
                               ld16(prow + kb * 32 + 16 + hi * 8));
    #pragma unroll
    for (int db = 0; db < 4; ++db) {
        v8f acc = zero8();
        const u16* vrow = vbase + (size_t)(db * 16 + m) * Tc;       // B lane: row d of V^T
        #pragma unroll
        for (int kb = 0; kb < 8; ++kb) {
            if (kb <= kbmax) {
                v16bf vb = make_frag(ld16(vrow + kb * 32 + hi * 8),
                                     ld16(vrow + kb * 32 + 16 + hi * 8));
                acc = wmma_bf16(pa[kb], vb, acc);
            }
        }
        u16* orow = o + ((size_t)b * Tc + r0) * (Hc * Dc) + h * Dc + db * 16 + m;
        #pragma unroll
        for (int r = 0; r < 8; ++r)
            orow[(size_t)(r + 8 * hi) * (Hc * Dc)] = f2bf(acc[r]);
    }
}

// ---------------- kernel 3: output projection y = o @ Wp + bp ----------------
__global__ __launch_bounds__(128) void proj_kernel(
    const u16* __restrict__ o, const u16* __restrict__ wpt,
    const float* __restrict__ bp, float* __restrict__ y)
{
    int cb = blockIdx.x % (Ec / 64);
    int rb = blockIdx.x / (Ec / 64);
    int lane = threadIdx.x & 31, wave = threadIdx.x >> 5;
    int m = lane & 15, hi = lane >> 4;
    int r0 = rb * 64 + wave * 16;

    const u16* orow = o + ((size_t)r0 + m) * Ec;
    v16bf a[12];
    #pragma unroll
    for (int kb = 0; kb < 12; ++kb)
        a[kb] = make_frag(ld16(orow + kb * 32 + hi * 8),
                          ld16(orow + kb * 32 + 16 + hi * 8));

    #pragma unroll
    for (int db = 0; db < 4; ++db) {
        int n = cb * 64 + db * 16 + m;
        const u16* wrow = wpt + (size_t)n * Ec;      // B lane: row n of Wp^T (column n of Wp)
        v8f acc = zero8();
        #pragma unroll
        for (int kb = 0; kb < 12; ++kb) {
            v16bf bfrag = make_frag(ld16(wrow + kb * 32 + hi * 8),
                                    ld16(wrow + kb * 32 + 16 + hi * 8));
            acc = wmma_bf16(a[kb], bfrag, acc);
        }
        float bias = bp[n];
        float* yrow = y + (size_t)r0 * Ec + n;
        #pragma unroll
        for (int r = 0; r < 8; ++r)
            yrow[(size_t)(r + 8 * hi) * Ec] = acc[r] + bias;
    }
}

// ---------------- host-side launcher ----------------
extern "C" void kernel_launch(void* const* d_in, const int* in_sizes, int n_in,
                              void* d_out, int out_size, void* d_ws, size_t ws_size,
                              hipStream_t stream) {
    const float* x  = (const float*)d_in[0];
    const float* Wq = (const float*)d_in[1];
    const float* bq = (const float*)d_in[2];
    const float* Wk = (const float*)d_in[3];
    const float* bk = (const float*)d_in[4];
    const float* Wv = (const float*)d_in[5];
    const float* bv = (const float*)d_in[6];
    const float* Wp = (const float*)d_in[7];
    const float* bp = (const float*)d_in[8];
    float* y = (float*)d_out;

    char* ws = (char*)d_ws;
    size_t p = 0;
    auto take = [&](size_t bytes) -> char* {
        char* r = ws + p;
        p = (p + bytes + 255) & ~(size_t)255;
        return r;
    };
    u16* wqt = (u16*)take((size_t)Hc * Dc * Ec * 2);
    u16* wkt = (u16*)take((size_t)Hc * Dc * Ec * 2);
    u16* wvt = (u16*)take((size_t)Hc * Dc * Ec * 2);
    u16* wpt = (u16*)take((size_t)Ec * Ec * 2);
    u16* qw  = (u16*)take((size_t)Bc * Hc * Tc * Dc * 2);
    u16* kw  = (u16*)take((size_t)Bc * Hc * Tc * Dc * 2);
    u16* vtw = (u16*)take((size_t)Bc * Hc * Dc * Tc * 2);
    u16* ow  = (u16*)take((size_t)Bc * Tc * Ec * 2);

    prep_weights<<<(Hc * Ec * Dc + 255) / 256, 256, 0, stream>>>(
        Wq, Wk, Wv, Wp, wqt, wkt, wvt, wpt);
    qkv_kernel<<<Bc * (Tc / TB), 128, 0, stream>>>(
        x, bq, bk, bv, wqt, wkt, wvt, qw, kw, vtw);
    attn_kernel<<<Bc * Hc * (Tc / 64), 128, 0, stream>>>(qw, kw, vtw, ow);
    proj_kernel<<<(Bc * Tc / 64) * (Ec / 64), 128, 0, stream>>>(ow, wpt, bp, y);
}
